// DecoderAttention_43061342109993
// MI455X (gfx1250) — matrix operand order repaired and checked by reading, not compiled, not run on hardware
//
#include <hip/hip_runtime.h>
#include <hip/hip_bf16.h>

#define D   256
#define H   4
#define DK  64
#define NV  40000
#define NE  320000
#define GB  64          // number of global nodes (B)
#define CAP_MAX 65536

typedef __attribute__((ext_vector_type(16))) _Float16 v16h;
typedef __attribute__((ext_vector_type(8)))  _Float16 v8h;
typedef __attribute__((ext_vector_type(8)))  float    v8f;

// ---------- WMMA fragment loaders (CDNA5 ISA layouts) ----------
// A matrix 16x32 f16: lane (row = lane&15, kh = lane>>4).
// elements 0..7  -> k = k0 + kh*8 + i
// elements 8..15 -> k = k0 + 16 + kh*8 + (i-8)
__device__ __forceinline__ v16h load_a_frag(const _Float16* row, int k0, int kh) {
  v8h lo = *(const v8h*)(row + k0 + kh * 8);
  v8h hi = *(const v8h*)(row + k0 + 16 + kh * 8);
  v16h r;
#pragma unroll
  for (int i = 0; i < 8; ++i) { r[i] = lo[i]; r[8 + i] = hi[i]; }
  return r;
}
// B matrix 32x16 f16: lane (col = lane&15, kh = lane>>4); elements 0..7 -> k = k0 + kh*8 + i, 8..15 -> k = k0 + 16 + kh*8 + (i-8)
// `col` points at row n of B^T (length-D contiguous).
__device__ __forceinline__ v16h load_b_frag(const _Float16* col, int k0, int kh) {
  v8h lo = *(const v8h*)(col + k0 + kh * 8);
  v8h hi = *(const v8h*)(col + k0 + 16 + kh * 8);
  v16h r;
#pragma unroll
  for (int i = 0; i < 8; ++i) { r[i] = lo[i]; r[8 + i] = hi[i]; }
  return r;
}

// ---------- ordered-float atomic-max keys ----------
__device__ __forceinline__ unsigned f2key(float f) {
  unsigned u = __float_as_uint(f);
  return (u & 0x80000000u) ? ~u : (u | 0x80000000u);
}
__device__ __forceinline__ float key2f(unsigned k) {
  unsigned u = (k & 0x80000000u) ? (k & 0x7FFFFFFFu) : ~k;
  return __uint_as_float(u);
}

// ---------- init all per-call state ----------
__global__ void k_init(int* cnt, unsigned* mkey, float* denom, float* outAcc) {
  int idx = blockIdx.x * blockDim.x + threadIdx.x;
  if (idx == 0) cnt[0] = 0;
  if (idx < GB * H) { mkey[idx] = 0u; denom[idx] = 0.f; }
  if (idx < GB * D) outAcc[idx] = 0.f;
}

// ---------- compact edges with dst < GB ----------
__global__ void k_filter(const long long* __restrict__ src,
                         const long long* __restrict__ dst,
                         int* cnt, int* esrc, int* edst, int cap) {
  int e = blockIdx.x * blockDim.x + threadIdx.x;
  if (e >= NE) return;
  long long d = dst[e];
  if (d < (long long)GB) {
    int pos = atomicAdd(cnt, 1);
    if (pos < cap) { esrc[pos] = (int)src[e]; edst[pos] = (int)d; }
  }
}

// ---------- Q = query @ WQ^T  (64x256, trivial) ----------
__global__ void k_q(const float* __restrict__ query, const float* __restrict__ WQ,
                    float* __restrict__ Qm) {
  int idx = blockIdx.x * blockDim.x + threadIdx.x;
  if (idx >= GB * D) return;
  int b = idx / D, d = idx % D;
  float acc = 0.f;
  for (int c = 0; c < D; ++c) acc += query[b * D + c] * WQ[d * D + c];
  Qm[idx] = acc;
}

// ---------- Pt[h][b][c] = sum_d Q[b, h*DK+d] * WK[h*DK+d, c]  (f16) ----------
__global__ void k_p(const float* __restrict__ Qm, const float* __restrict__ WK,
                    _Float16* __restrict__ Pt) {
  int idx = blockIdx.x * blockDim.x + threadIdx.x;
  if (idx >= H * GB * D) return;
  int h = idx / (GB * D);
  int rem = idx % (GB * D);
  int b = rem / D, c = rem % D;
  float acc = 0.f;
  for (int d = 0; d < DK; ++d)
    acc += Qm[b * D + h * DK + d] * WK[(h * DK + d) * D + c];
  Pt[idx] = (_Float16)acc;
}

// ---------- f16 copies of WV / WO ----------
__global__ void k_convert(const float* __restrict__ WV, const float* __restrict__ WO,
                          _Float16* __restrict__ WVh, _Float16* __restrict__ WOh) {
  int idx = blockIdx.x * blockDim.x + threadIdx.x;
  if (idx >= D * D) return;
  WVh[idx] = (_Float16)WV[idx];
  WOh[idx] = (_Float16)WO[idx];
}

// ---------- per-16-edge-tile scores: S = Xg(16xD) @ P_h(DxGB), pick dst column ----------
__global__ __launch_bounds__(32) void k_scores(const float* __restrict__ x,
                                               const int* __restrict__ cnt,
                                               const int* __restrict__ esrc,
                                               const int* __restrict__ edst,
                                               const _Float16* __restrict__ Pt,
                                               float* __restrict__ sArr, int cap) {
  int EC = min(*cnt, cap);
  int tileBase = blockIdx.x * 16;
  if (tileBase >= EC) return;

  __shared__ _Float16 Xg[16][D];
  __shared__ int dloc[16];
  int lane = threadIdx.x;

  if (lane < 16) {
    int e = tileBase + lane;
    dloc[lane] = (e < EC) ? edst[e] : -1;
  }
  for (int i = lane; i < 16 * (D / 4); i += 32) {
    int m = i / (D / 4), c4 = i % (D / 4);
    int e = tileBase + m;
    if (e < EC) {
      float4 v = ((const float4*)(x + (size_t)esrc[e] * D))[c4];
      Xg[m][c4 * 4 + 0] = (_Float16)v.x;
      Xg[m][c4 * 4 + 1] = (_Float16)v.y;
      Xg[m][c4 * 4 + 2] = (_Float16)v.z;
      Xg[m][c4 * 4 + 3] = (_Float16)v.w;
    } else {
      Xg[m][c4 * 4 + 0] = (_Float16)0.f; Xg[m][c4 * 4 + 1] = (_Float16)0.f;
      Xg[m][c4 * 4 + 2] = (_Float16)0.f; Xg[m][c4 * 4 + 3] = (_Float16)0.f;
    }
  }
  __syncthreads();

  int row = lane & 15, kh = lane >> 4;
  for (int h = 0; h < H; ++h) {
    const _Float16* Ph = Pt + (size_t)h * GB * D;
    for (int n0 = 0; n0 < GB; n0 += 16) {
      v8f acc = {};
      const _Float16* bcol = Ph + (size_t)(n0 + row) * D;
#pragma unroll
      for (int k0 = 0; k0 < D; k0 += 32) {
        v16h a = load_a_frag(&Xg[row][0], k0, kh);
        v16h b = load_b_frag(bcol, k0, kh);
        acc = __builtin_amdgcn_wmma_f32_16x16x32_f16(false, a, false, b,
                                                     (short)0, acc, false, false);
      }
      int n = n0 + (lane & 15);
      int mb = (lane >> 4) * 8;
#pragma unroll
      for (int r = 0; r < 8; ++r) {
        int m = mb + r;
        if (dloc[m] == n)
          sArr[(size_t)(tileBase + m) * H + h] = acc[r] * 0.125f;  // 1/sqrt(DK)
      }
    }
  }
}

// ---------- segment max ----------
__global__ void k_max(const int* __restrict__ cnt, const int* __restrict__ edst,
                      const float* __restrict__ sArr, unsigned* mkey, int cap) {
  int EC = min(*cnt, cap);
  int idx = blockIdx.x * blockDim.x + threadIdx.x;
  if (idx >= EC * H) return;
  int e = idx / H, h = idx % H;
  atomicMax(&mkey[edst[e] * H + h], f2key(sArr[idx]));
}

// ---------- exp numerators + segment sum ----------
__global__ void k_expdenom(const int* __restrict__ cnt, const int* __restrict__ edst,
                           const float* __restrict__ sArr, const unsigned* __restrict__ mkey,
                           float* __restrict__ anum, float* denom, int cap) {
  int EC = min(*cnt, cap);
  int idx = blockIdx.x * blockDim.x + threadIdx.x;
  if (idx >= EC * H) return;
  int e = idx / H, h = idx % H;
  float mv = key2f(mkey[edst[e] * H + h]);
  float ev = expf(sArr[idx] - mv);
  anum[idx] = ev;
  atomicAdd(&denom[edst[e] * H + h], ev);
}

// ---------- alpha = num / denom ----------
__global__ void k_alpha(const int* __restrict__ cnt, const int* __restrict__ edst,
                        const float* __restrict__ anum, const float* __restrict__ denom,
                        float* __restrict__ alphaArr, int cap) {
  int EC = min(*cnt, cap);
  int idx = blockIdx.x * blockDim.x + threadIdx.x;
  if (idx >= EC * H) return;
  int e = idx / H, h = idx % H;
  alphaArr[idx] = anum[idx] / denom[edst[e] * H + h];
}

// ---------- V projection per edge tile + weighted scatter into outAcc ----------
__global__ __launch_bounds__(32) void k_aggregate(const float* __restrict__ x,
                                                  const int* __restrict__ cnt,
                                                  const int* __restrict__ esrc,
                                                  const int* __restrict__ edst,
                                                  const _Float16* __restrict__ WVh,
                                                  const float* __restrict__ alphaArr,
                                                  float* outAcc, int cap) {
  int EC = min(*cnt, cap);
  int tileBase = blockIdx.x * 16;
  if (tileBase >= EC) return;

  __shared__ _Float16 Xg[16][D];
  __shared__ int dloc[16];
  __shared__ float aLoc[16][H];
  int lane = threadIdx.x;

  if (lane < 16) {
    int e = tileBase + lane;
    dloc[lane] = (e < EC) ? edst[e] : -1;
  }
  for (int i = lane; i < 16 * H; i += 32) {
    int m = i >> 2, h = i & 3;
    int e = tileBase + m;
    aLoc[m][h] = (e < EC) ? alphaArr[(size_t)e * H + h] : 0.f;
  }
  for (int i = lane; i < 16 * (D / 4); i += 32) {
    int m = i / (D / 4), c4 = i % (D / 4);
    int e = tileBase + m;
    if (e < EC) {
      float4 v = ((const float4*)(x + (size_t)esrc[e] * D))[c4];
      Xg[m][c4 * 4 + 0] = (_Float16)v.x;
      Xg[m][c4 * 4 + 1] = (_Float16)v.y;
      Xg[m][c4 * 4 + 2] = (_Float16)v.z;
      Xg[m][c4 * 4 + 3] = (_Float16)v.w;
    } else {
      Xg[m][c4 * 4 + 0] = (_Float16)0.f; Xg[m][c4 * 4 + 1] = (_Float16)0.f;
      Xg[m][c4 * 4 + 2] = (_Float16)0.f; Xg[m][c4 * 4 + 3] = (_Float16)0.f;
    }
  }
  __syncthreads();

  int row = lane & 15, kh = lane >> 4;
  for (int h = 0; h < H; ++h) {
    for (int n0 = 0; n0 < DK; n0 += 16) {
      v8f acc = {};
      const _Float16* bcol = WVh + (size_t)(h * DK + n0 + row) * D;  // row of WV == col of WV^T
#pragma unroll
      for (int k0 = 0; k0 < D; k0 += 32) {
        v16h a = load_a_frag(&Xg[row][0], k0, kh);
        v16h b = load_b_frag(bcol, k0, kh);
        acc = __builtin_amdgcn_wmma_f32_16x16x32_f16(false, a, false, b,
                                                     (short)0, acc, false, false);
      }
      int n = n0 + (lane & 15);
      int mb = (lane >> 4) * 8;
#pragma unroll
      for (int r = 0; r < 8; ++r) {
        int m = mb + r;
        int dn = dloc[m];
        if (dn >= 0)
          atomicAdd(&outAcc[(size_t)dn * D + h * DK + n], aLoc[m][h] * acc[r]);
      }
    }
  }
}

// ---------- final projection: r = outAcc(64xD) @ WO^T ----------
__global__ __launch_bounds__(32) void k_output(const float* __restrict__ outAcc,
                                               const _Float16* __restrict__ WOh,
                                               float* __restrict__ rOut) {
  int mt = blockIdx.x >> 4;   // 0..3 (rows 16*mt..)
  int nt = blockIdx.x & 15;   // 0..15 (cols 16*nt..)
  __shared__ _Float16 Ag[16][D];
  int lane = threadIdx.x;
  for (int i = lane; i < 16 * (D / 4); i += 32) {
    int m = i / (D / 4), c4 = i % (D / 4);
    float4 v = ((const float4*)(outAcc + (size_t)(mt * 16 + m) * D))[c4];
    Ag[m][c4 * 4 + 0] = (_Float16)v.x;
    Ag[m][c4 * 4 + 1] = (_Float16)v.y;
    Ag[m][c4 * 4 + 2] = (_Float16)v.z;
    Ag[m][c4 * 4 + 3] = (_Float16)v.w;
  }
  __syncthreads();

  int row = lane & 15, kh = lane >> 4;
  v8f acc = {};
  const _Float16* bcol = WOh + (size_t)(nt * 16 + row) * D;  // row o of WO == col o of WO^T
#pragma unroll
  for (int k0 = 0; k0 < D; k0 += 32) {
    v16h a = load_a_frag(&Ag[row][0], k0, kh);
    v16h b = load_b_frag(bcol, k0, kh);
    acc = __builtin_amdgcn_wmma_f32_16x16x32_f16(false, a, false, b,
                                                 (short)0, acc, false, false);
  }
  int n = nt * 16 + (lane & 15);
  int mb = (lane >> 4) * 8;
#pragma unroll
  for (int r = 0; r < 8; ++r)
    rOut[(size_t)(mt * 16 + mb + r) * D + n] = acc[r];
}

extern "C" void kernel_launch(void* const* d_in, const int* in_sizes, int n_in,
                              void* d_out, int out_size, void* d_ws, size_t ws_size,
                              hipStream_t stream) {
  (void)in_sizes; (void)n_in; (void)out_size;
  const float*      query = (const float*)d_in[0];
  const float*      x     = (const float*)d_in[1];
  const float*      WQ    = (const float*)d_in[2];
  const float*      WK    = (const float*)d_in[3];
  const float*      WV    = (const float*)d_in[4];
  const float*      WO    = (const float*)d_in[5];
  const long long*  src   = (const long long*)d_in[6];
  const long long*  dst   = (const long long*)d_in[7];
  float* out = (float*)d_out;

  // workspace layout
  char* w = (char*)d_ws;
  int*       cnt    = (int*)w;                                      // 256 B header
  float*     Qm     = (float*)(w + 256);                            // 65536 B
  _Float16*  Pt     = (_Float16*)(w + 256 + 65536);                 // 131072 B
  _Float16*  WVh    = (_Float16*)(w + 256 + 65536 + 131072);        // 131072 B
  _Float16*  WOh    = (_Float16*)(w + 256 + 65536 + 2 * 131072);    // 131072 B
  unsigned*  mkey   = (unsigned*)(w + 256 + 65536 + 3 * 131072);    // 1024 B
  float*     denom  = (float*)(w + 256 + 65536 + 3 * 131072 + 1024);// 1024 B
  float*     outAcc = (float*)(w + 256 + 65536 + 3 * 131072 + 2048);// 65536 B
  const size_t fixedBytes = 256 + 65536 + 3 * 131072 + 2048 + 65536; // 526592

  int cap = 16;
  if (ws_size > fixedBytes + 16 * 56) {
    long long c = (long long)(ws_size - fixedBytes) / 56;
    if (c > CAP_MAX) c = CAP_MAX;
    c = (c / 16) * 16;
    if (c >= 16) cap = (int)c;
  }
  int*   esrc     = (int*)(w + fixedBytes);
  int*   edst     = esrc + cap;
  float* sArr     = (float*)(edst + cap);
  float* anum     = sArr + (size_t)cap * H;
  float* alphaArr = anum + (size_t)cap * H;

  k_init   <<<(GB * D + 255) / 256, 256, 0, stream>>>(cnt, mkey, denom, outAcc);
  k_filter <<<(NE + 255) / 256, 256, 0, stream>>>(src, dst, cnt, esrc, edst, cap);
  k_q      <<<(GB * D + 255) / 256, 256, 0, stream>>>(query, WQ, Qm);
  k_p      <<<(H * GB * D + 255) / 256, 256, 0, stream>>>(Qm, WK, Pt);
  k_convert<<<(D * D + 255) / 256, 256, 0, stream>>>(WV, WO, WVh, WOh);

  int tiles = cap / 16;
  k_scores   <<<tiles, 32, 0, stream>>>(x, cnt, esrc, edst, Pt, sArr, cap);
  int eh = (cap * H + 255) / 256;
  k_max      <<<eh, 256, 0, stream>>>(cnt, edst, sArr, mkey, cap);
  k_expdenom <<<eh, 256, 0, stream>>>(cnt, edst, sArr, mkey, anum, denom, cap);
  k_alpha    <<<eh, 256, 0, stream>>>(cnt, edst, anum, denom, alphaArr, cap);
  k_aggregate<<<tiles, 32, 0, stream>>>(x, cnt, esrc, edst, WVh, alphaArr, outAcc, cap);
  k_output   <<<64, 32, 0, stream>>>(outAcc, WOh, out);
}
